// SelfAttentionLM_51651276701831
// MI455X (gfx1250) — matrix operand order repaired and checked by reading, not compiled, not run on hardware
//
#include <hip/hip_runtime.h>
#include <hip/hip_bf16.h>

#define VOCAB 50257
#define EMB   16
#define BS    8
#define MCW   512
#define SEQ   (BS * MCW)              // 4096 rows total
#define NT    ((VOCAB + 15) / 16)     // 3142 column tiles
#define NPAD  (NT * 16)               // 50272 padded vocab

typedef __attribute__((ext_vector_type(16))) _Float16 v16h;
typedef __attribute__((ext_vector_type(8)))  _Float16 v8h;
typedef __attribute__((ext_vector_type(8)))  float    v8f;

__device__ __forceinline__ float wave_max32(float x) {
  #pragma unroll
  for (int off = 16; off > 0; off >>= 1)
    x = fmaxf(x, __shfl_xor(x, off, 32));
  return x;
}
__device__ __forceinline__ float wave_sum32(float x) {
  #pragma unroll
  for (int off = 16; off > 0; off >>= 1)
    x += __shfl_xor(x, off, 32);
  return x;
}

// ---------------------------------------------------------------------------
// Kernel 1: embedding gather + Q/K/V projections (one thread per token)
// q[j] = bq[j] + sum_i e[i] * Wq[j,i]   (embs @ Wq.T + bq)
// ---------------------------------------------------------------------------
__global__ void qkv_kernel(const int* __restrict__ x,
                           const float* __restrict__ emb,
                           const float* __restrict__ Wq, const float* __restrict__ bq,
                           const float* __restrict__ Wk, const float* __restrict__ bk,
                           const float* __restrict__ Wv, const float* __restrict__ bv,
                           float* __restrict__ q, float* __restrict__ k,
                           float* __restrict__ v) {
  int tid = blockIdx.x * blockDim.x + threadIdx.x;
  if (tid >= SEQ) return;
  int tok = x[tid];
  float e[EMB];
  #pragma unroll
  for (int i = 0; i < EMB; ++i) e[i] = emb[tok * EMB + i];
  #pragma unroll
  for (int j = 0; j < EMB; ++j) {
    float sq = bq[j], sk = bk[j], sv = bv[j];
    #pragma unroll
    for (int i = 0; i < EMB; ++i) {
      sq = fmaf(e[i], Wq[j * EMB + i], sq);
      sk = fmaf(e[i], Wk[j * EMB + i], sk);
      sv = fmaf(e[i], Wv[j * EMB + i], sv);
    }
    q[tid * EMB + j] = sq;
    k[tid * EMB + j] = sk;
    v[tid * EMB + j] = sv;
  }
}

// ---------------------------------------------------------------------------
// Kernel 2: per-(batch, key-column) softmax stats over the QUERY dim.
// Column kc is unmasked for q in [kc, MCW).  Exact-zero scores -> -inf (skip).
// One wave per column; lanes stride the q range; shuffle-reduce max then sum.
// ---------------------------------------------------------------------------
__global__ void colstats_kernel(const float* __restrict__ q,
                                const float* __restrict__ k,
                                float* __restrict__ colmax,
                                float* __restrict__ denom) {
  int wid  = (blockIdx.x * blockDim.x + threadIdx.x) >> 5;
  int lane = threadIdx.x & 31;
  if (wid >= SEQ) return;
  int b = wid / MCW, kc = wid % MCW;
  float kv[EMB];
  #pragma unroll
  for (int i = 0; i < EMB; ++i) kv[i] = k[wid * EMB + i];
  const float* qb = q + b * MCW * EMB;

  float m = -INFINITY;
  for (int qi = kc + lane; qi < MCW; qi += 32) {
    float s = 0.0f;
    #pragma unroll
    for (int i = 0; i < EMB; ++i) s = fmaf(qb[qi * EMB + i], kv[i], s);
    if (s != 0.0f) m = fmaxf(m, s);
  }
  m = wave_max32(m);

  float d = 0.0f;
  for (int qi = kc + lane; qi < MCW; qi += 32) {
    float s = 0.0f;
    #pragma unroll
    for (int i = 0; i < EMB; ++i) s = fmaf(qb[qi * EMB + i], kv[i], s);
    if (s != 0.0f) d += __expf(s - m);
  }
  d = wave_sum32(d);

  if (lane == 0) { colmax[wid] = m; denom[wid] = d; }
}

// ---------------------------------------------------------------------------
// Kernel 3: out[b,w,:] = sum_{p<=w, s!=0} (exp(s - colmax[p]) / denom[p]) * V[b,p,:]
// One wave per output row; lanes stride p; butterfly-reduce 16 accumulators.
// Writes the attention output directly in f16 (WMMA A operand).
// ---------------------------------------------------------------------------
__global__ void attnout_kernel(const float* __restrict__ q,
                               const float* __restrict__ k,
                               const float* __restrict__ v,
                               const float* __restrict__ colmax,
                               const float* __restrict__ denom,
                               _Float16* __restrict__ outh) {
  int wid  = (blockIdx.x * blockDim.x + threadIdx.x) >> 5;
  int lane = threadIdx.x & 31;
  if (wid >= SEQ) return;
  int b = wid / MCW, w = wid % MCW;
  float qr[EMB];
  #pragma unroll
  for (int i = 0; i < EMB; ++i) qr[i] = q[wid * EMB + i];
  const float* kb = k + b * MCW * EMB;
  const float* vb = v + b * MCW * EMB;
  const float* cm = colmax + b * MCW;
  const float* dn = denom + b * MCW;

  float acc[EMB];
  #pragma unroll
  for (int i = 0; i < EMB; ++i) acc[i] = 0.0f;

  for (int p = lane; p <= w; p += 32) {
    float s = 0.0f;
    #pragma unroll
    for (int i = 0; i < EMB; ++i) s = fmaf(qr[i], kb[p * EMB + i], s);
    if (s == 0.0f) continue;                       // exact-zero quirk -> -inf -> weight 0
    float wgt = __expf(s - cm[p]) / dn[p];
    #pragma unroll
    for (int i = 0; i < EMB; ++i) acc[i] = fmaf(wgt, vb[p * EMB + i], acc[i]);
  }
  #pragma unroll
  for (int i = 0; i < EMB; ++i) acc[i] = wave_sum32(acc[i]);

  if (lane == 0) {
    #pragma unroll
    for (int i = 0; i < EMB; ++i) outh[wid * EMB + i] = (_Float16)acc[i];
  }
}

// ---------------------------------------------------------------------------
// Kernel 4: Wl (VOCAB x 16 f32) -> f16, padded to NPAD rows (pad rows = 0)
// ---------------------------------------------------------------------------
__global__ void cvt_wl_kernel(const float* __restrict__ Wl,
                              _Float16* __restrict__ wlh) {
  int tid = blockIdx.x * blockDim.x + threadIdx.x;
  if (tid >= NPAD * EMB) return;
  int n = tid / EMB;
  _Float16 val = (_Float16)0.0f;
  if (n < VOCAB) val = (_Float16)Wl[tid];
  wlh[tid] = val;
}

// ---------------------------------------------------------------------------
// Kernel 5: logits = out @ Wl^T + bl via v_wmma_f32_16x16x32_f16.
// One wave per 16x16 output tile. K=16 real data, zero-padded to K=32.
// grid = (ceil(NT/8), SEQ/16), block = 256 (8 waves); wave w -> N-tile
// blockIdx.x*8+w, M-tile blockIdx.y.  Non-temporal stores (823MB > L2).
// ---------------------------------------------------------------------------
__global__ void logits_wmma_kernel(const _Float16* __restrict__ outh,
                                   const _Float16* __restrict__ wlh,
                                   const float* __restrict__ bl,
                                   float* __restrict__ logits) {
  int wave = threadIdx.x >> 5;
  int lane = threadIdx.x & 31;
  int nt = blockIdx.x * 8 + wave;
  int mt = blockIdx.y;
  if (nt >= NT) return;                 // wave-uniform exit: EXEC stays all-ones
  int m_base = mt * 16;
  int n_base = nt * 16;

  // A operand: 16x32 f16.  Lanes 0-15: M=lane, elems 0..15 = K0..15;
  // lanes 16-31: K16..31 = 0.
  int mrow = m_base + (lane & 15);
  v16h a = {};
  if (lane < 16) a = *(const v16h*)(outh + mrow * EMB);

  // B operand: 32x16 f16, B[k][n] = Wl[n][k].  Lanes 0-15 hold K=0..15 for
  // column n_base+lane (one contiguous 32B Wl row); lanes 16-31 = K16..31 = 0.
  v16h bmat = {};
  if (lane < 16) bmat = *(const v16h*)(wlh + (n_base + lane) * EMB);

  // C/D: VGPR j, lanes 0-15 -> (M=j, N=lane); lanes 16-31 -> (M=j+8, N=lane-16).
  int col = n_base + (lane & 15);
  float blv = 0.0f;
  if (col < VOCAB) blv = bl[col];
  v8f c = {};
  #pragma unroll
  for (int j = 0; j < 8; ++j) c[j] = blv;

  c = __builtin_amdgcn_wmma_f32_16x16x32_f16(
      /*neg_a=*/false, a, /*neg_b=*/false, bmat,
      /*c_mod=*/(short)0, c, /*reuse_a=*/false, /*reuse_b=*/false);

  int rbase = m_base + ((lane >> 4) << 3);
  if (col < VOCAB) {
    #pragma unroll
    for (int j = 0; j < 8; ++j) {
      __builtin_nontemporal_store(c[j],
          logits + (size_t)(rbase + j) * (size_t)VOCAB + (size_t)col);
    }
  }
}

// ---------------------------------------------------------------------------
extern "C" void kernel_launch(void* const* d_in, const int* in_sizes, int n_in,
                              void* d_out, int out_size, void* d_ws, size_t ws_size,
                              hipStream_t stream) {
  const int*   x   = (const int*)d_in[0];
  const float* emb = (const float*)d_in[1];
  const float* Wq  = (const float*)d_in[2];
  const float* bq  = (const float*)d_in[3];
  const float* Wk  = (const float*)d_in[4];
  const float* bk  = (const float*)d_in[5];
  const float* Wv  = (const float*)d_in[6];
  const float* bv  = (const float*)d_in[7];
  const float* Wl  = (const float*)d_in[8];
  const float* bl  = (const float*)d_in[9];
  float* logits = (float*)d_out;

  // Workspace layout (bytes):
  char* ws = (char*)d_ws;
  float*    q      = (float*)(ws + 0);          // 4096*16*4 = 262144
  float*    k      = (float*)(ws + 262144);     // 262144
  float*    v      = (float*)(ws + 524288);     // 262144
  float*    colmax = (float*)(ws + 786432);     // 16384
  float*    denom  = (float*)(ws + 802816);     // 16384
  _Float16* outh   = (_Float16*)(ws + 819200);  // 4096*16*2  = 131072
  _Float16* wlh    = (_Float16*)(ws + 950272);  // 50272*16*2 = 1608704  (total ~2.6MB)

  qkv_kernel<<<SEQ / 256, 256, 0, stream>>>(x, emb, Wq, bq, Wk, bk, Wv, bv, q, k, v);
  colstats_kernel<<<(SEQ * 32) / 256, 256, 0, stream>>>(q, k, colmax, denom);
  cvt_wl_kernel<<<(NPAD * EMB + 255) / 256, 256, 0, stream>>>(Wl, wlh);
  attnout_kernel<<<(SEQ * 32) / 256, 256, 0, stream>>>(q, k, v, colmax, denom, outh);

  dim3 grid((NT + 7) / 8, SEQ / 16);            // (393, 256)
  logits_wmma_kernel<<<grid, 256, 0, stream>>>(outh, wlh, bl, logits);
}
